// Attention_with_ALiBi_Bias_85933705658792
// MI455X (gfx1250) — matrix operand order repaired and checked by reading, not compiled, not run on hardware
//
#include <hip/hip_runtime.h>
#include <hip/hip_bf16.h>

// ---------------------------------------------------------------------------
// Attention with ALiBi bias, MI455X (gfx1250, wave32, WMMA + async-LDS DMA).
// Pipeline:
//   1) Q = x@Wq, K = x@Wk, V = x@Wv      (bf16 WMMA GEMM, bf16 results in ws)
//   2) flash-attention per (b,h,128-row) (bf16 WMMA for QK^T and P@V,
//                                         fp32 online softmax + ALiBi + causal,
//                                         K tile staged via async global->LDS)
//   3) out = attn@Wo + bo                (bf16 WMMA GEMM, async A staging)
// ---------------------------------------------------------------------------

typedef __bf16 v16bf __attribute__((ext_vector_type(16)));
typedef float  v8f   __attribute__((ext_vector_type(8)));

union Frag {
  v16bf v;
  uint4 q[2];
};

__device__ __forceinline__ unsigned short f32_to_bf16_bits(float f) {
  unsigned int u = __builtin_bit_cast(unsigned int, f);
  u += 0x7FFFu + ((u >> 16) & 1u);   // round-to-nearest-even
  return (unsigned short)(u >> 16);
}

// ---- async global->LDS (ASYNCcnt) path, hedged on builtin availability ----
#if defined(__HIP_DEVICE_COMPILE__) && \
    __has_builtin(__builtin_amdgcn_global_load_async_to_lds_b128)
#define HAVE_ASYNC_LDS 1
#else
#define HAVE_ASYNC_LDS 0
#endif

// exact parameter element type per hipcc diagnostic:
//   __attribute__((__vector_size__(4 * sizeof(int)))) int
typedef int v4i_vs __attribute__((__vector_size__(4 * sizeof(int))));
#if HAVE_ASYNC_LDS
typedef __attribute__((address_space(1))) v4i_vs* gbl_v4i_p;
typedef __attribute__((address_space(3))) v4i_vs* lds_v4i_p;
#endif

__device__ __forceinline__ void async_copy16(unsigned short* lds_dst,
                                             const unsigned short* gsrc) {
#if HAVE_ASYNC_LDS
  __builtin_amdgcn_global_load_async_to_lds_b128(
      (gbl_v4i_p)gsrc, (lds_v4i_p)lds_dst, /*offset=*/0, /*cpol=*/0);
#else
  *(uint4*)lds_dst = *(const uint4*)gsrc;   // synchronous fallback
#endif
}

__device__ __forceinline__ void wait_async0() {
#if HAVE_ASYNC_LDS
#if __has_builtin(__builtin_amdgcn_s_wait_asynccnt)
  __builtin_amdgcn_s_wait_asynccnt(0);
#else
  asm volatile("s_wait_asynccnt 0x0" ::: "memory");
#endif
#endif
}

// ---------------------------------------------------------------------------
// Tiled GEMM:  C[M,Nc] = A[M,K] @ W[K,Nc]   (W fp32 row-major, converted)
// Block tile 64(M) x 128(N), 8 waves in 2x4 grid, each wave 32x32 (2x2 WMMA),
// K-step 32.  A fragments: runs at k = 8*half and 16+8*half.  B stored
// transposed in LDS (Bs[n][k]) so B fragments, like A, are runs at
// k = 8*half and 16+8*half.  When A is already bf16 the A tile is staged with async
// global->LDS DMA (one b128 per thread).
// ---------------------------------------------------------------------------
template <bool A_BF16, bool OUT_F32_BIAS>
__global__ __launch_bounds__(256) void gemm_bf16_wmma(
    const void* __restrict__ Ain, const float* __restrict__ W,
    const float* __restrict__ bias, void* __restrict__ Cout,
    int M, int K, int Nc) {
  __shared__ __align__(16) unsigned short As[64 * 48];   //  6 KB, padded rows
  __shared__ __align__(16) unsigned short Bs[128 * 48];  // 12 KB, padded rows

  const int tid  = threadIdx.x;
  const int lane = tid & 31;
  const int l15  = lane & 15;
  const int half = lane >> 4;
  const int w    = tid >> 5;   // wave 0..7
  const int wm   = w >> 2;     // 0..1  (M sub-tile of 32)
  const int wn   = w & 3;      // 0..3  (N sub-tile of 32)
  const int m0   = blockIdx.y * 64;
  const int n0   = blockIdx.x * 128;

  v8f acc[2][2];
  acc[0][0] = (v8f)0.0f; acc[0][1] = (v8f)0.0f;
  acc[1][0] = (v8f)0.0f; acc[1][1] = (v8f)0.0f;

  const float*          Af = (const float*)Ain;
  const unsigned short* Ab = (const unsigned short*)Ain;

  for (int k0 = 0; k0 < K; k0 += 32) {
    __syncthreads();
    // hint the next K-step's weight tile toward the caches
    if (k0 + 32 < K)
      __builtin_prefetch(&W[(size_t)(k0 + 32 + (tid >> 7)) * Nc + n0 + (tid & 127)], 0, 1);

    // stage A tile 64x32
    if (A_BF16) {
      // one 16-byte async DMA per thread covers the whole 64x32 bf16 tile
      int m = tid >> 2, kk0 = (tid & 3) * 8;
      async_copy16(&As[m * 48 + kk0],
                   Ab + (size_t)(m0 + m) * K + (size_t)(k0 + kk0));
    } else {
#pragma unroll
      for (int s = 0; s < 8; ++s) {
        int e = tid + 256 * s;          // 0..2047
        int m = e >> 5, kk = e & 31;
        size_t g = (size_t)(m0 + m) * K + (size_t)(k0 + kk);
        As[m * 48 + kk] = f32_to_bf16_bits(Af[g]);
      }
    }
    // stage B tile 32x128, transposed into Bs[n][k] (fp32->bf16 convert)
#pragma unroll
    for (int s = 0; s < 16; ++s) {
      int e = tid + 256 * s;          // 0..4095
      int n = e & 127, k = e >> 7;
      Bs[n * 48 + k] = f32_to_bf16_bits(W[(size_t)(k0 + k) * Nc + (n0 + n)]);
    }
    if (A_BF16) wait_async0();
    __syncthreads();

    Frag a[2], b[2];
#pragma unroll
    for (int im = 0; im < 2; ++im) {
      int m = wm * 32 + im * 16 + l15;
      a[im].q[0] = *(const uint4*)&As[m * 48 + 8 * half];
      a[im].q[1] = *(const uint4*)&As[m * 48 + 16 + 8 * half];
    }
#pragma unroll
    for (int in = 0; in < 2; ++in) {
      int n = wn * 32 + in * 16 + l15;
      b[in].q[0] = *(const uint4*)&Bs[n * 48 + 8 * half];
      b[in].q[1] = *(const uint4*)&Bs[n * 48 + 16 + 8 * half];
    }
#pragma unroll
    for (int im = 0; im < 2; ++im)
#pragma unroll
      for (int in = 0; in < 2; ++in)
        acc[im][in] = __builtin_amdgcn_wmma_f32_16x16x32_bf16(
            false, a[im].v, false, b[in].v, (short)0, acc[im][in], false, false);
  }

  // writeout: C layout row m = r + 8*half, col n = lane&15
#pragma unroll
  for (int im = 0; im < 2; ++im)
#pragma unroll
    for (int in = 0; in < 2; ++in)
#pragma unroll
      for (int r = 0; r < 8; ++r) {
        int mg = m0 + wm * 32 + im * 16 + r + 8 * half;
        int ng = n0 + wn * 32 + in * 16 + l15;
        float v = acc[im][in][r];
        if (OUT_F32_BIAS)
          ((float*)Cout)[(size_t)mg * Nc + ng] = v + bias[ng];
        else
          ((unsigned short*)Cout)[(size_t)mg * Nc + ng] = f32_to_bf16_bits(v);
      }
}

// ---------------------------------------------------------------------------
// Flash attention with ALiBi + causal mask.
// Grid: (N/128, H, B).  Block: 256 threads = 8 waves; wave w owns query rows
// i0 + 16w .. i0 + 16w + 15.  Keys processed in blocks of 32; the K tile is
// staged with async global->LDS DMA, V is transposed through VGPRs.
// ---------------------------------------------------------------------------
__global__ __launch_bounds__(256) void flash_alibi_attn(
    const unsigned short* __restrict__ Qb, const unsigned short* __restrict__ Kb,
    const unsigned short* __restrict__ Vb, unsigned short* __restrict__ Ob) {
  constexpr int N = 2048, H = 16, DH = 64, HD = 1024;
  __shared__ __align__(16) unsigned short Ks[32 * 64];       // K block, row-major
  __shared__ __align__(16) unsigned short VTs[64 * 32];      // V block, transposed
  __shared__ __align__(16) unsigned short Ps[8 * 16 * 32];   // per-wave P staging

  const int tid  = threadIdx.x;
  const int lane = tid & 31;
  const int l15  = lane & 15;
  const int half = lane >> 4;
  const int w    = tid >> 5;
  const int i0   = blockIdx.x * 128;
  const int h    = blockIdx.y;
  const int b    = blockIdx.z;

  const float scale = 0.125f;  // DH^-0.5
  const float slope = exp2f(-8.0f * (float)(h + 1) / (float)H);

  // Q fragments (A layout) straight from global, kept in registers.
  const int iq = i0 + w * 16 + l15;
  const unsigned short* qrow = Qb + ((size_t)(b * N + iq)) * HD + h * DH;
  Frag qf[2];
#pragma unroll
  for (int t = 0; t < 2; ++t) {
    qf[t].q[0] = *(const uint4*)(qrow + 32 * t + 8 * half);
    qf[t].q[1] = *(const uint4*)(qrow + 32 * t + 16 + 8 * half);
  }

  v8f o[4];
  o[0] = (v8f)0.0f; o[1] = (v8f)0.0f; o[2] = (v8f)0.0f; o[3] = (v8f)0.0f;
  float mi[8], li[8];
#pragma unroll
  for (int r = 0; r < 8; ++r) { mi[r] = -1e30f; li[r] = 0.0f; }

  const int njb = i0 / 32 + 4;  // causal bound for this block's last row
  unsigned short* Pw = &Ps[w * 16 * 32];

  for (int jbi = 0; jbi < njb; ++jbi) {
    const int jb = jbi * 32;
    __syncthreads();
    // K tile (32x64 bf16 = 4 KB): one 16-byte async DMA per thread
    {
      int jj = tid >> 3, dd0 = (tid & 7) * 8;
      async_copy16(&Ks[jj * 64 + dd0],
                   Kb + ((size_t)(b * N + jb + jj)) * HD + h * DH + dd0);
      if (jbi + 1 < njb) {
        __builtin_prefetch(Kb + ((size_t)(b * N + jb + 32 + jj)) * HD + h * DH + dd0, 0, 1);
        __builtin_prefetch(Vb + ((size_t)(b * N + jb + 32 + jj)) * HD + h * DH + dd0, 0, 1);
      }
    }
    // V tile, transposed into VTs (data rearrangement -> through VGPRs)
#pragma unroll
    for (int s = 0; s < 8; ++s) {
      int e  = tid + 256 * s;   // 0..2047
      int jj = e >> 6, dd = e & 63;
      VTs[dd * 32 + jj] = Vb[((size_t)(b * N + jb + jj)) * HD + h * DH + dd];
    }
    wait_async0();
    __syncthreads();

    // S = Q @ K^T  (16x32 per wave: two 16-col tiles, two K-steps over d)
    v8f s0 = (v8f)0.0f, s1 = (v8f)0.0f;
#pragma unroll
    for (int t = 0; t < 2; ++t) {
      Frag kb0, kb1;
      kb0.q[0] = *(const uint4*)&Ks[l15 * 64 + 32 * t + 8 * half];
      kb0.q[1] = *(const uint4*)&Ks[l15 * 64 + 32 * t + 16 + 8 * half];
      kb1.q[0] = *(const uint4*)&Ks[(16 + l15) * 64 + 32 * t + 8 * half];
      kb1.q[1] = *(const uint4*)&Ks[(16 + l15) * 64 + 32 * t + 16 + 8 * half];
      s0 = __builtin_amdgcn_wmma_f32_16x16x32_bf16(false, qf[t].v, false, kb0.v,
                                                   (short)0, s0, false, false);
      s1 = __builtin_amdgcn_wmma_f32_16x16x32_bf16(false, qf[t].v, false, kb1.v,
                                                   (short)0, s1, false, false);
    }

    // online softmax with ALiBi bias + causal mask (fp32, C layout)
#pragma unroll
    for (int r = 0; r < 8; ++r) {
      const int irow = i0 + w * 16 + r + 8 * half;
      const int j0 = jb + l15;
      const int j1 = jb + 16 + l15;
      float a0 = s0[r] * scale + slope * (float)(irow - j0);
      float a1 = s1[r] * scale + slope * (float)(irow - j1);
      if (j0 > irow) a0 = -1e30f;
      if (j1 > irow) a1 = -1e30f;
      float m = fmaxf(a0, a1);
#pragma unroll
      for (int off = 1; off < 16; off <<= 1)
        m = fmaxf(m, __shfl_xor(m, off, 32));
      float mnew  = fmaxf(mi[r], m);
      float alpha = __expf(mi[r] - mnew);
      float p0 = __expf(a0 - mnew);
      float p1 = __expf(a1 - mnew);
      float rs = p0 + p1;
#pragma unroll
      for (int off = 1; off < 16; off <<= 1)
        rs += __shfl_xor(rs, off, 32);
      li[r] = li[r] * alpha + rs;
      mi[r] = mnew;
      o[0][r] *= alpha; o[1][r] *= alpha; o[2][r] *= alpha; o[3][r] *= alpha;
      // C layout -> A layout via per-wave LDS staging (bf16)
      Pw[(r + 8 * half) * 32 + l15]      = f32_to_bf16_bits(p0);
      Pw[(r + 8 * half) * 32 + 16 + l15] = f32_to_bf16_bits(p1);
    }
    __syncthreads();

    // O += P @ V   (k = 32 keys, four 16-col d tiles)
    Frag pa;
    pa.q[0] = *(const uint4*)&Pw[l15 * 32 + 8 * half];
    pa.q[1] = *(const uint4*)&Pw[l15 * 32 + 16 + 8 * half];
#pragma unroll
    for (int td = 0; td < 4; ++td) {
      Frag vb;
      vb.q[0] = *(const uint4*)&VTs[(td * 16 + l15) * 32 + 8 * half];
      vb.q[1] = *(const uint4*)&VTs[(td * 16 + l15) * 32 + 16 + 8 * half];
      o[td] = __builtin_amdgcn_wmma_f32_16x16x32_bf16(false, pa.v, false, vb.v,
                                                      (short)0, o[td], false, false);
    }
  }

  // normalize and store attention output (bf16 for the final projection)
#pragma unroll
  for (int td = 0; td < 4; ++td)
#pragma unroll
    for (int r = 0; r < 8; ++r) {
      int i = i0 + w * 16 + r + 8 * half;
      int d = td * 16 + l15;
      Ob[((size_t)(b * N + i)) * HD + h * DH + d] =
          f32_to_bf16_bits(o[td][r] / li[r]);
    }
}

// ---------------------------------------------------------------------------
extern "C" void kernel_launch(void* const* d_in, const int* in_sizes, int n_in,
                              void* d_out, int out_size, void* d_ws,
                              size_t ws_size, hipStream_t stream) {
  (void)in_sizes; (void)n_in; (void)out_size; (void)ws_size;
  const float* x  = (const float*)d_in[0];
  const float* Wq = (const float*)d_in[1];
  const float* Wk = (const float*)d_in[2];
  const float* Wv = (const float*)d_in[3];
  const float* Wo = (const float*)d_in[4];
  const float* bo = (const float*)d_in[5];
  float* out = (float*)d_out;

  const int Bn = 2, Nn = 2048, Dn = 1024, Hn = 16, HDn = 1024;
  const int M = Bn * Nn;  // 4096
  size_t qkvBytes = (size_t)M * HDn * sizeof(unsigned short);  // 8 MB each
  unsigned short* Qb = (unsigned short*)d_ws;
  unsigned short* Kb = (unsigned short*)((char*)d_ws + qkvBytes);
  unsigned short* Vb = (unsigned short*)((char*)d_ws + 2 * qkvBytes);
  unsigned short* Ab = (unsigned short*)((char*)d_ws + 3 * qkvBytes);

  dim3 blk(256);
  dim3 gProj(HDn / 128, M / 64);  // 8 x 64
  gemm_bf16_wmma<false, false><<<gProj, blk, 0, stream>>>(x, Wq, nullptr, Qb, M, Dn, HDn);
  gemm_bf16_wmma<false, false><<<gProj, blk, 0, stream>>>(x, Wk, nullptr, Kb, M, Dn, HDn);
  gemm_bf16_wmma<false, false><<<gProj, blk, 0, stream>>>(x, Wv, nullptr, Vb, M, Dn, HDn);

  dim3 gAttn(Nn / 128, Hn, Bn);   // 16 x 16 x 2 = 512 blocks
  flash_alibi_attn<<<gAttn, blk, 0, stream>>>(Qb, Kb, Vb, Ab);

  dim3 gOut(Dn / 128, M / 64);    // 8 x 64
  gemm_bf16_wmma<true, true><<<gOut, blk, 0, stream>>>(Ab, Wo, bo, out, M, HDn, Dn);
}
